// SynthesisBlock_28346784153760
// MI455X (gfx1250) — compile-verified, matrix operand order checked by reading, not run on hardware
//
#include <hip/hip_runtime.h>
#include <hip/hip_bf16.h>

typedef __attribute__((ext_vector_type(16))) _Float16 v16h;
typedef __attribute__((ext_vector_type(8)))  _Float16 v8h;
typedef __attribute__((ext_vector_type(8)))  float    v8f;

#define B_   16
#define H_   64
#define W_   64
#define HW_  4096          // 64*64
#define CIN  256
#define CO1  256
#define CO2  512
#define WDIM 128
#define KTOT 2304          // 9 * CIN

// ---------------------------------------------------------------------------
// Small prep kernels (negligible cost vs the convs)
// ---------------------------------------------------------------------------

// Ksq[i*co + o] = sum over 9 taps of K[tap][i][o]^2   (K is [3,3,CIN,co])
__global__ void ksq_kernel(const float* __restrict__ K, float* __restrict__ out, int co) {
    int idx = blockIdx.x * blockDim.x + threadIdx.x;
    if (idx >= CIN * co) return;
    float s = 0.f;
#pragma unroll
    for (int t = 0; t < 9; ++t) {
        float v = K[(size_t)t * CIN * co + idx];
        s += v * v;
    }
    out[idx] = s;
}

// s[b*CIN + c] = w[b] @ A[:,c] + bias[c]
__global__ void style_kernel(const float* __restrict__ w, const float* __restrict__ A,
                             const float* __restrict__ bias, float* __restrict__ s) {
    int idx = blockIdx.x * blockDim.x + threadIdx.x;   // b*CIN + c
    if (idx >= B_ * CIN) return;
    int b = idx >> 8, c = idx & 255;
    float acc = bias[c];
    const float* wb = w + b * WDIM;
#pragma unroll 4
    for (int j = 0; j < WDIM; ++j) acc += wb[j] * A[j * CIN + c];
    s[idx] = acc;
}

// d[b*co + o] = rsqrt( sum_i s[b,i]^2 * Ksq[i,o] + 1e-8 )
__global__ void demod_kernel(const float* __restrict__ s, const float* __restrict__ ksq,
                             float* __restrict__ d, int co) {
    int idx = blockIdx.x * blockDim.x + threadIdx.x;   // b*co + o
    if (idx >= B_ * co) return;
    int b = idx / co, o = idx % co;
    float acc = 1e-8f;
    const float* sb = s + b * CIN;
#pragma unroll 4
    for (int i = 0; i < CIN; ++i) {
        float si = sb[i];
        acc += si * si * ksq[i * co + o];
    }
    d[idx] = rsqrtf(acc);
}

// xm[b,p,i] = (f16)(x[b,p,i] * s1[b,i])   — vectorized by 4
__global__ void modx_kernel(const float* __restrict__ x, const float* __restrict__ s1,
                            _Float16* __restrict__ xm) {
    int idx = blockIdx.x * blockDim.x + threadIdx.x;
    int e = idx * 4;
    if (e >= B_ * HW_ * CIN) return;
    int c = e & 255;
    int b = e >> 20;                 // e / (HW_*CIN); HW_*CIN = 2^20
    float4 v = *(const float4*)(x + e);
    const float* sb = s1 + b * CIN + c;
    xm[e + 0] = (_Float16)(v.x * sb[0]);
    xm[e + 1] = (_Float16)(v.y * sb[1]);
    xm[e + 2] = (_Float16)(v.z * sb[2]);
    xm[e + 3] = (_Float16)(v.w * sb[3]);
}

// Kt[o][t*CIN + i] = (f16) K[t][i][o]  — transpose to out-channel-major for B-frags
__global__ void kconv_kernel(const float* __restrict__ K, _Float16* __restrict__ Kt, int co) {
    int idx = blockIdx.x * blockDim.x + threadIdx.x;   // t*CIN*co + i*co + o
    if (idx >= 9 * CIN * co) return;
    int o = idx % co;
    int rest = idx / co;
    int i = rest % CIN;
    int t = rest / CIN;
    Kt[(size_t)o * KTOT + t * CIN + i] = (_Float16)K[idx];
}

// ---------------------------------------------------------------------------
// Implicit-GEMM modulated conv via v_wmma_f32_16x16x32_f16
//   M = HW (pixels), N = cout, K = 9*CIN (im2col over 3x3 taps)
//   Block: 256 threads = 8 waves; block tile 64(M) x 128(N) = one image row.
//   Waves arranged 4(M) x 2(N); each wave: 16x64 strip, 4 accumulators,
//   A-fragment reused across 4 WMMAs per k-step.
//   mode 0: epilogue -> d1, noise1, bias1, lrelu, pre-modulate by s2, f16 out
//   mode 1: epilogue -> d2, noise2, bias2, lrelu, pixel-shuffle scatter, f32 out
// ---------------------------------------------------------------------------
__global__ __launch_bounds__(256)
void conv_wmma_kernel(const _Float16* __restrict__ Xm,  // [B][HW][CIN] f16 (pre-modulated)
                      const _Float16* __restrict__ Kt,  // [cout][KTOT] f16
                      const float* __restrict__ dmod,   // [B][cout]
                      const float* __restrict__ noise,  // [B][HW]
                      const float* __restrict__ nscale, // [1]
                      const float* __restrict__ bias,   // [cout]
                      const float* __restrict__ s_next, // [B][CIN] (mode 0 only)
                      _Float16* __restrict__ outH,      // mode 0
                      float* __restrict__ outF,         // mode 1
                      int cout, int mode) {
    const int tid  = threadIdx.x;
    const int lane = tid & 31;
    const int wave = tid >> 5;
    const int b      = blockIdx.z;
    const int h      = blockIdx.x;       // M tile == image row h (W_ == 64)
    const int n_base = blockIdx.y * 128;
    const int wm = (wave & 3) * 16;      // wave M offset within 64
    const int wn = (wave >> 2) * 64;     // wave N offset within 128

    // stride 80 halfs = 160 B: every 8-half segment stays 16B-aligned, banks skewed
    __shared__ __align__(16) _Float16 sA[64][80];
    __shared__ __align__(16) _Float16 sB[128][80];

    v8f acc[4] = {};

    const _Float16* XmB = Xm + ((size_t)b * HW_ << 8);

    // per-thread hoisted B base pointers (2 chunks of the 128x64 tile each... 4 chunks)
    // B tile: 128 rows x 8 segs = 1024 chunks of 16B; 4 per thread
    // A tile:  64 rows x 8 segs =  512 chunks of 16B; 2 per thread

    for (int tap = 0; tap < 9; ++tap) {
        const int dh = tap / 3 - 1, dw = tap % 3 - 1;
        const int hs = h + dh;
        const bool hvalid = (unsigned)hs < (unsigned)H_;   // uniform over block
        const _Float16* rowA = XmB + ((size_t)(hs * W_) << 8);
        const _Float16* srcB = Kt + (size_t)tap * CIN;

        for (int kc = 0; kc < CIN; kc += 64) {
            __syncthreads();   // previous stage's LDS reads complete
            // ---- stage A tile 64x64 (2 chunks/thread)
#pragma unroll
            for (int c = tid; c < 512; c += 256) {
                int row = c >> 3;            // pixel w within the image row
                int seg = (c & 7) * 8;
                int wsrc = row + dw;
                int4 av = make_int4(0, 0, 0, 0);
                if (hvalid && (unsigned)wsrc < (unsigned)W_)
                    av = *(const int4*)(rowA + ((size_t)wsrc << 8) + kc + seg);
                *(int4*)&sA[row][seg] = av;
            }
            // ---- stage B tile 128x64 (4 chunks/thread)
#pragma unroll
            for (int c = tid; c < 1024; c += 256) {
                int row = c >> 3;
                int seg = (c & 7) * 8;
                *(int4*)&sB[row][seg] =
                    *(const int4*)(srcB + (size_t)(n_base + row) * KTOT + kc + seg);
            }
            // prefetch next K-chunk of weights while we compute (global_prefetch_b8)
            if (kc + 64 < CIN)
                __builtin_prefetch(srcB + (size_t)(n_base + (tid >> 1)) * KTOT + kc + 64, 0, 1);
            __syncthreads();   // LDS tiles visible

#pragma unroll
            for (int kk = 0; kk < 64; kk += 32) {
                // ISA 16-bit fragment layout: lanes 0-15 hold K {0..7,16..23},
                // lanes 16-31 hold K {8..15,24..31} (relative to kk)
                const int kb = kk + ((lane >= 16) ? 8 : 0);
                const int fr = lane & 15;
                v8h a0 = *(const v8h*)&sA[wm + fr][kb];
                v8h a1 = *(const v8h*)&sA[wm + fr][kb + 16];
                v16h af = __builtin_shufflevector(a0, a1,
                          0, 1, 2, 3, 4, 5, 6, 7, 8, 9, 10, 11, 12, 13, 14, 15);
#pragma unroll
                for (int j = 0; j < 4; ++j) {
                    v8h b0 = *(const v8h*)&sB[wn + j * 16 + fr][kb];
                    v8h b1 = *(const v8h*)&sB[wn + j * 16 + fr][kb + 16];
                    v16h bf = __builtin_shufflevector(b0, b1,
                              0, 1, 2, 3, 4, 5, 6, 7, 8, 9, 10, 11, 12, 13, 14, 15);
                    acc[j] = __builtin_amdgcn_wmma_f32_16x16x32_f16(
                        false, af, false, bf, (short)0, acc[j], false, false);
                }
            }
        }
    }

    // ---- epilogue: C/D layout -> VGPR r holds M=r (lanes 0-15) / M=r+8 (16-31), N=lane%16
    const int fr    = lane & 15;
    const int mrow0 = h * W_ + wm + ((lane >= 16) ? 8 : 0);
    const float ns  = nscale[0];

    if (mode == 0) {
#pragma unroll
        for (int j = 0; j < 4; ++j) {
            const int o = n_base + wn + j * 16 + fr;
            const float dm  = dmod[b * cout + o];
            const float bi  = bias[o];
            const float s2v = s_next[b * CIN + o];
#pragma unroll
            for (int r = 0; r < 8; ++r) {
                int pp = mrow0 + r;
                float v = acc[j][r] * dm + noise[b * HW_ + pp] * ns + bi;
                v = (v > 0.f) ? v : 0.2f * v;
                outH[(((size_t)b * HW_ + pp) << 8) + o] = (_Float16)(v * s2v);
            }
        }
    } else {
#pragma unroll
        for (int j = 0; j < 4; ++j) {
            const int o = n_base + wn + j * 16 + fr;
            const float dm = dmod[b * cout + o];
            const float bi = bias[o];
            const int co = o & 127, sx = (o >> 7) & 1, sy = (o >> 8) & 1;
#pragma unroll
            for (int r = 0; r < 8; ++r) {
                int pp = mrow0 + r;
                float v = acc[j][r] * dm + noise[b * HW_ + pp] * ns + bi;
                v = (v > 0.f) ? v : 0.2f * v;
                int h2 = ((pp >> 6) << 1) + sy;
                int w2 = ((pp & 63) << 1) + sx;
                outF[(((size_t)b * 128 + h2) * 128 + w2) * 128 + co] = v;
            }
        }
    }
}

// ---------------------------------------------------------------------------
extern "C" void kernel_launch(void* const* d_in, const int* in_sizes, int n_in,
                              void* d_out, int out_size, void* d_ws, size_t ws_size,
                              hipStream_t stream) {
    const float* x    = (const float*)d_in[0];
    const float* w    = (const float*)d_in[1];
    const float* a1w  = (const float*)d_in[2];
    const float* a1b  = (const float*)d_in[3];
    const float* k1   = (const float*)d_in[4];
    const float* ns1  = (const float*)d_in[5];
    const float* b1   = (const float*)d_in[6];
    const float* a2w  = (const float*)d_in[7];
    const float* a2b  = (const float*)d_in[8];
    const float* k2   = (const float*)d_in[9];
    const float* ns2  = (const float*)d_in[10];
    const float* b2   = (const float*)d_in[11];
    const float* n1   = (const float*)d_in[12];
    const float* n2   = (const float*)d_in[13];
    float* out = (float*)d_out;

    char* ws = (char*)d_ws;
    float*    s1   = (float*)(ws + 0);          // 16*256*4     = 16384
    float*    s2   = (float*)(ws + 16384);      // 16384
    float*    d1   = (float*)(ws + 32768);      // 16384
    float*    d2   = (float*)(ws + 49152);      // 16*512*4     = 32768
    float*    ksq1 = (float*)(ws + 81920);      // 256*256*4    = 262144
    float*    ksq2 = (float*)(ws + 344064);     // 256*512*4    = 524288
    _Float16* k1t  = (_Float16*)(ws + 868352);  // 256*2304*2   = 1179648
    _Float16* k2t  = (_Float16*)(ws + 2048000); // 512*2304*2   = 2359296
    _Float16* xm   = (_Float16*)(ws + 4407296); // 16*4096*256*2 = 33554432
    _Float16* hm   = (_Float16*)(ws + 37961728);// 33554432  (end ~71.5 MB)

    // 1) tap-summed squared weights (for demod)
    ksq_kernel<<<(CIN * CO1 + 255) / 256, 256, 0, stream>>>(k1, ksq1, CO1);
    ksq_kernel<<<(CIN * CO2 + 255) / 256, 256, 0, stream>>>(k2, ksq2, CO2);
    // 2) styles
    style_kernel<<<(B_ * CIN + 255) / 256, 256, 0, stream>>>(w, a1w, a1b, s1);
    style_kernel<<<(B_ * CIN + 255) / 256, 256, 0, stream>>>(w, a2w, a2b, s2);
    // 3) demod factors
    demod_kernel<<<(B_ * CO1 + 255) / 256, 256, 0, stream>>>(s1, ksq1, d1, CO1);
    demod_kernel<<<(B_ * CO2 + 255) / 256, 256, 0, stream>>>(s2, ksq2, d2, CO2);
    // 4) pre-modulate x by s1, convert to f16
    modx_kernel<<<(B_ * HW_ * CIN / 4 + 255) / 256, 256, 0, stream>>>(x, s1, xm);
    // 5) weights -> f16, [cout][9*CIN]
    kconv_kernel<<<(9 * CIN * CO1 + 255) / 256, 256, 0, stream>>>(k1, k1t, CO1);
    kconv_kernel<<<(9 * CIN * CO2 + 255) / 256, 256, 0, stream>>>(k2, k2t, CO2);

    // 6) conv1 (WMMA): [B,HW,CIN] x [CO1,KTOT] -> hm (f16, pre-modulated by s2)
    {
        dim3 grid(H_, CO1 / 128, B_);
        conv_wmma_kernel<<<grid, 256, 0, stream>>>(xm, k1t, d1, n1, ns1, b1, s2,
                                                   hm, nullptr, CO1, 0);
    }
    // 7) conv2 (WMMA) + pixel-shuffle scatter into d_out
    {
        dim3 grid(H_, CO2 / 128, B_);
        conv_wmma_kernel<<<grid, 256, 0, stream>>>(hm, k2t, d2, n2, ns2, b2, nullptr,
                                                   nullptr, out, CO2, 1);
    }
    (void)in_sizes; (void)n_in; (void)out_size; (void)ws_size;
}
